// DecoderLSTMCell_30666066493836
// MI455X (gfx1250) — compile-verified, matrix operand order checked by reading, not run on hardware
//
#include <hip/hip_runtime.h>
#include <hip/hip_bf16.h>

// ---------------- types for CDNA5 WMMA ----------------
typedef __attribute__((ext_vector_type(16))) __bf16 v16bf;
typedef __attribute__((ext_vector_type(8)))  float  v8f;

// ---------------- constants ----------------
#define BATCH 256
#define TCAP  40
#define TSEQ  41
#define FDIM  2048
#define EDIM  1024
#define HDIM  1024
#define GDIM  4096   // 4*H
#define VOCAB 12000
#define BM    64      // rows per block
#define BN    128     // cols per block
#define KT    64      // K per staging step
#define LDAP  72      // padded LDS stride (bf16 elems) = 144B, 16B-aligned
#define LDBP  72

// ---------------- bf16 helpers (bit-exact RNE) ----------------
__device__ inline __bf16 f2bf(float f) {
  union { float f; unsigned u; } v; v.f = f;
  unsigned r = (v.u + 0x7FFFu + ((v.u >> 16) & 1u)) >> 16;
  unsigned short s = (unsigned short)r;
  __bf16 out; __builtin_memcpy(&out, &s, 2);
  return out;
}

// ---------------- f32 -> bf16 bulk convert (vectorized) ----------------
__global__ __launch_bounds__(256)
void cvt_bf16_kernel(const float* __restrict__ src, __bf16* __restrict__ dst, long n) {
  long i = ((long)blockIdx.x * 256 + threadIdx.x) * 4;
  if (i + 3 < n) {
    float4 v = *(const float4*)(src + i);
    dst[i + 0] = f2bf(v.x);
    dst[i + 1] = f2bf(v.y);
    dst[i + 2] = f2bf(v.z);
    dst[i + 3] = f2bf(v.w);
  }
}

// ---------------- meta: packed-sequence offsets ----------------
// meta[0..40]=off[t]; meta[41]=off[T]; meta[42]=N_total; meta[43]=bs_last
__global__ void meta_kernel(const int* __restrict__ lengths, int* __restrict__ meta) {
  if (threadIdx.x == 0 && blockIdx.x == 0) {
    int off = 0;
    for (int t = 0; t < TSEQ; ++t) {
      meta[t] = off;
      int bs = 0;
      for (int b = 0; b < BATCH; ++b) bs += (lengths[b] > t) ? 1 : 0;
      off += bs;
    }
    meta[41] = off;
    meta[42] = off;
    int bl = 0;
    for (int b = 0; b < BATCH; ++b) bl += (lengths[b] == TSEQ) ? 1 : 0;
    meta[43] = bl;
  }
}

// ---------------- fold gate biases once ----------------
__global__ void bias_sum_kernel(const float* __restrict__ bih,
                                const float* __restrict__ bhh,
                                float* __restrict__ bsum) {
  int i = blockIdx.x * 256 + threadIdx.x;
  if (i < GDIM) bsum[i] = bih[i] + bhh[i];
}

// ---------------- zero initial state ----------------
__global__ void init_state_kernel(float* __restrict__ h, float* __restrict__ c,
                                  __bf16* __restrict__ hb) {
  int i = blockIdx.x * 256 + threadIdx.x;   // B*H threads
  h[i] = 0.f; c[i] = 0.f; hb[i] = f2bf(0.f);
}

// ---------------- gather word embeddings -> bf16 X rows t=1..40 ----------
__global__ void gather_embed_kernel(const int* __restrict__ caps,
                                    const float* __restrict__ table,
                                    __bf16* __restrict__ Xb) {
  long idx = (long)blockIdx.x * 256 + threadIdx.x;   // over B*TCAP*E
  int  e   = (int)(idx & (EDIM - 1));
  long r   = idx >> 10;                              // b*TCAP + tt
  int  b   = (int)(r / TCAP);
  int  tt  = (int)(r % TCAP);
  int  tok = caps[b * TCAP + tt];
  float v  = table[(size_t)tok * EDIM + e];
  Xb[((size_t)(tt + 1) * BATCH + b) * EDIM + e] = f2bf(v);
}

// ---------------- fused WMMA GEMM ------------------------------------
// C[M,N] = [A0 | A1] (M x K) * [B0 | B1]^T (N x K) + bias
// K in [0,Ksplit) reads A0/B0 (row len Ksplit); K in [Ksplit,K) reads A1/B1
// (row len K-Ksplit). A1/B1 may be null when Ksplit==K.
// Block: 8 waves, tile 64x128. Waves 2(M) x 4(N); each wave owns 32x32
// (2x2 subtiles of 16x16 -> 4 accumulators, 8 WMMAs per K-step of 64,
// A/B fragments each reused twice).
__global__ __launch_bounds__(256)
void gemm_wmma_kernel(const __bf16* __restrict__ A0, const __bf16* __restrict__ A1,
                      const __bf16* __restrict__ B0, const __bf16* __restrict__ B1,
                      int Ksplit,
                      const float* __restrict__ bias,
                      float* __restrict__ outF, __bf16* __restrict__ outB,
                      int M, int N, int K, const int* __restrict__ Mdev) {
  const int Meff = Mdev ? Mdev[0] : M;     // uniform scalar load
  const int tid  = threadIdx.x;
  const int wave = tid >> 5;
  const int lane = tid & 31;
  const int r0   = blockIdx.x * BM;
  if (r0 >= Meff) return;
  const int nBlk = blockIdx.y * BN;
  const int wm   = wave >> 2;              // 0..1 (M half: 32 rows)
  const int wn   = wave & 3;               // 0..3 (N quarter: 32 cols)
  const int K1   = K - Ksplit;             // second-source row length

  __shared__ __bf16 As[BM * LDAP];         // 64 x 64 (padded)  9KB
  __shared__ __bf16 Bs[BN * LDBP];         // 128 x 64 (padded) 18KB

  v8f acc00 = {0.f,0.f,0.f,0.f,0.f,0.f,0.f,0.f};
  v8f acc01 = acc00, acc10 = acc00, acc11 = acc00;

  // ---- staging coordinates (fixed per thread) ----
  const int aRow = tid >> 2;               // 0..63
  const int aK   = (tid & 3) * 16;         // 0,16,32,48 (two 16B chunks)
  const int bRow = tid >> 1;               // 0..127
  const int bK   = (tid & 1) * 32;         // 0 or 32 (two 16B chunks)
  const int gr   = r0 + aRow;
  const int gn   = nBlk + bRow;
  const bool aOk = (gr < Meff);
  const bool bOk = (gn < N);
  // hoisted source base pointers (64-bit mul once, K-loop adds only)
  const __bf16* aBase0 = A0 + (size_t)gr * Ksplit + aK;
  const __bf16* aBase1 = A1 ? (A1 + (size_t)gr * K1 + aK) : aBase0;
  const __bf16* bBase0 = B0 + (size_t)gn * Ksplit + bK;
  const __bf16* bBase1 = B1 ? (B1 + (size_t)gn * K1 + bK) : bBase0;

  // ---- fragment coordinates ----
  const int kb  = (lane >> 4) << 3;        // 0 or 8 per ISA 16-bit layout
  const int ma0 = (wm << 5) + (lane & 15); // M subtile 0 row
  const int ma1 = ma0 + 16;                // M subtile 1 row
  const int nl0 = (wn << 5) + (lane & 15); // N subtile 0 col
  const int nl1 = nl0 + 16;                // N subtile 1 col

  for (int k0 = 0; k0 < K; k0 += KT) {
    const bool first = (k0 < Ksplit);
    const __bf16* aSrc = (first ? aBase0 : aBase1) + (first ? k0 : (k0 - Ksplit));
    const __bf16* bSrc = (first ? bBase0 : bBase1) + (first ? k0 : (k0 - Ksplit));

    // ---- stage A: 64x64 bf16, two 16B chunks per thread ----
    {
      uint4 v0 = {0u,0u,0u,0u}, v1 = {0u,0u,0u,0u};
      if (aOk) {
        const uint4* src = (const uint4*)aSrc;
        v0 = src[0]; v1 = src[1];
      }
      uint4* dst = (uint4*)&As[aRow * LDAP + aK];
      dst[0] = v0; dst[1] = v1;
    }
    // ---- stage B: 128x64 bf16, two 16B chunks per thread ----
    {
      uint4 v0 = {0u,0u,0u,0u}, v1 = {0u,0u,0u,0u};
      if (bOk) {
        const uint4* src = (const uint4*)bSrc;
        if (k0 + KT < K) __builtin_prefetch((const void*)(src + 8), 0, 0);
        v0 = src[0]; v1 = src[1];
      }
      uint4* dst = (uint4*)&Bs[bRow * LDBP + bK];
      dst[0] = v0; dst[1] = v1;
    }
    __syncthreads();

    // ---- 2 K-chunks x (2 A frags x 2 B frags) -> 8 WMMAs per wave ----
#pragma unroll
    for (int kc = 0; kc < KT; kc += 32) {
      const __bf16* ap0 = &As[ma0 * LDAP + kc + kb];
      const __bf16* ap1 = &As[ma1 * LDAP + kc + kb];
      const __bf16* bp0 = &Bs[nl0 * LDBP + kc + kb];
      const __bf16* bp1 = &Bs[nl1 * LDBP + kc + kb];
      v16bf af0, af1, bf0, bf1;
#pragma unroll
      for (int j = 0; j < 8; ++j) {
        af0[j] = ap0[j]; af0[j + 8] = ap0[16 + j];
        af1[j] = ap1[j]; af1[j + 8] = ap1[16 + j];
        bf0[j] = bp0[j]; bf0[j + 8] = bp0[16 + j];
        bf1[j] = bp1[j]; bf1[j + 8] = bp1[16 + j];
      }
      acc00 = __builtin_amdgcn_wmma_f32_16x16x32_bf16(false, af0, false, bf0,
                                                      (short)0, acc00, false, false);
      acc01 = __builtin_amdgcn_wmma_f32_16x16x32_bf16(false, af0, false, bf1,
                                                      (short)0, acc01, false, false);
      acc10 = __builtin_amdgcn_wmma_f32_16x16x32_bf16(false, af1, false, bf0,
                                                      (short)0, acc10, false, false);
      acc11 = __builtin_amdgcn_wmma_f32_16x16x32_bf16(false, af1, false, bf1,
                                                      (short)0, acc11, false, false);
    }
    __syncthreads();
  }

  // ---- epilogue: VGPR i -> row (subtile base) + i + 8*(lane>=16) ----
  const int rAdd = (lane >> 4) << 3;
  const int rb0  = r0 + (wm << 5) + rAdd;  // M subtile 0
  const int rb1  = rb0 + 16;               // M subtile 1
  const int c0   = nBlk + (wn << 5) + (lane & 15);
  const int c1   = c0 + 16;
  const float bv0 = (bias && c0 < N) ? bias[c0] : 0.f;
  const float bv1 = (bias && c1 < N) ? bias[c1] : 0.f;
#pragma unroll
  for (int i = 0; i < 8; ++i) {
    const int r0i = rb0 + i, r1i = rb1 + i;
    if (r0i < Meff) {
      if (c0 < N) {
        float v = acc00[i] + bv0;
        size_t o = (size_t)r0i * N + c0;
        if (outF) outF[o] = v;
        if (outB) outB[o] = f2bf(v);
      }
      if (c1 < N) {
        float v = acc01[i] + bv1;
        size_t o = (size_t)r0i * N + c1;
        if (outF) outF[o] = v;
        if (outB) outB[o] = f2bf(v);
      }
    }
    if (r1i < Meff) {
      if (c0 < N) {
        float v = acc10[i] + bv0;
        size_t o = (size_t)r1i * N + c0;
        if (outF) outF[o] = v;
        if (outB) outB[o] = f2bf(v);
      }
      if (c1 < N) {
        float v = acc11[i] + bv1;
        size_t o = (size_t)r1i * N + c1;
        if (outF) outF[o] = v;
        if (outB) outB[o] = f2bf(v);
      }
    }
  }
}

// ---------------- LSTM cell + packed scatter ----------------
__global__ __launch_bounds__(256)
void lstm_cell_kernel(const float* __restrict__ gates,
                      float* __restrict__ h, float* __restrict__ c,
                      __bf16* __restrict__ hb, __bf16* __restrict__ hsb,
                      const int* __restrict__ lengths,
                      const int* __restrict__ meta,
                      float* __restrict__ dout, int t) {
  int idx = blockIdx.x * 256 + threadIdx.x;   // B*H
  int b = idx >> 10;
  int j = idx & (HDIM - 1);
  const int Nt = meta[42];
  const int bsLast = meta[43];

  size_t gbase = (size_t)b * GDIM;
  float gi = gates[gbase + j];
  float gf = gates[gbase + HDIM + j];
  float gg = gates[gbase + 2 * HDIM + j];
  float go = gates[gbase + 3 * HDIM + j];
  gi = 1.f / (1.f + __expf(-gi));
  gf = 1.f / (1.f + __expf(-gf));
  go = 1.f / (1.f + __expf(-go));
  gg = tanhf(gg);

  float cOld = c[idx], hOld = h[idx];
  float cNew = gf * cOld + gi * gg;
  float hNew = go * tanhf(cNew);
  bool  m    = (t < lengths[b]);
  float hW   = m ? hNew : hOld;
  float cW   = m ? cNew : cOld;

  h[idx] = hW; c[idx] = cW; hb[idx] = f2bf(hW);

  if (m) {
    size_t pr = (size_t)meta[t] + b;                            // packed row
    size_t hsBase = (size_t)Nt * VOCAB + 2ull * bsLast * HDIM;  // h_states
    dout[hsBase + pr * HDIM + j] = hW;
    dout[hsBase + (size_t)Nt * HDIM + pr * HDIM + j] = cW;      // c_states
    hsb[pr * HDIM + j] = f2bf(hW);
  }
  if (t == TSEQ - 1 && lengths[b] == TSEQ) {                    // finals
    size_t fb = (size_t)Nt * VOCAB;
    dout[fb + (size_t)b * HDIM + j] = hW;                       // h_fin
    dout[fb + (size_t)bsLast * HDIM + (size_t)b * HDIM + j] = cW; // c_fin
  }
}

// ---------------- launcher ----------------
extern "C" void kernel_launch(void* const* d_in, const int* in_sizes, int n_in,
                              void* d_out, int out_size, void* d_ws, size_t ws_size,
                              hipStream_t stream) {
  const float* features = (const float*)d_in[0];   // B x F
  const int*   caps     = (const int*)d_in[1];     // B x TCAP
  const int*   lengths  = (const int*)d_in[2];     // B
  const float* visual_W = (const float*)d_in[3];   // E x F
  const float* visual_b = (const float*)d_in[4];   // E
  const float* embed    = (const float*)d_in[5];   // V x E
  const float* W_ih     = (const float*)d_in[6];   // 4H x E
  const float* W_hh     = (const float*)d_in[7];   // 4H x H
  const float* b_ih     = (const float*)d_in[8];   // 4H
  const float* b_hh     = (const float*)d_in[9];   // 4H
  const float* out_W    = (const float*)d_in[10];  // V x H
  const float* out_b    = (const float*)d_in[11];  // V
  float* dout = (float*)d_out;

  // workspace carve-up (256B aligned chunks)
  char* ws = (char*)d_ws;
  size_t o = 0;
  auto take = [&](size_t bytes) {
    char* p = ws + o;
    o += (bytes + 255) & ~(size_t)255;
    return p;
  };
  __bf16* Xb    = (__bf16*)take((size_t)TSEQ * BATCH * EDIM * 2);  // 21.5 MB
  __bf16* hb    = (__bf16*)take((size_t)BATCH * HDIM * 2);
  __bf16* hsb   = (__bf16*)take((size_t)TSEQ * BATCH * HDIM * 2);  // packed h bf16
  float*  hcur  = (float*)take((size_t)BATCH * HDIM * 4);
  float*  ccur  = (float*)take((size_t)BATCH * HDIM * 4);
  float*  gates = (float*)take((size_t)BATCH * GDIM * 4);
  float*  bsum  = (float*)take((size_t)GDIM * 4);
  int*    meta  = (int*)take(64 * 4);
  __bf16* featb = (__bf16*)take((size_t)BATCH * FDIM * 2);         // 1 MB
  __bf16* visWb = (__bf16*)take((size_t)EDIM * FDIM * 2);          // 4 MB
  __bf16* Wib   = (__bf16*)take((size_t)GDIM * EDIM * 2);          // 8 MB
  __bf16* Whb   = (__bf16*)take((size_t)GDIM * HDIM * 2);          // 8 MB
  __bf16* outWb = (__bf16*)take((size_t)VOCAB * HDIM * 2);         // 24 MB
  (void)ws_size; (void)in_sizes; (void)n_in; (void)out_size;

  // 1. one-time weight/feature bf16 conversion (read once, reused many times)
  auto cvt = [&](const float* s, __bf16* d, long n) {
    hipLaunchKernelGGL(cvt_bf16_kernel, dim3((unsigned)((n / 4 + 255) / 256)),
                       dim3(256), 0, stream, s, d, n);
  };
  cvt(features, featb, (long)BATCH * FDIM);
  cvt(visual_W, visWb, (long)EDIM * FDIM);
  cvt(W_ih, Wib, (long)GDIM * EDIM);
  cvt(W_hh, Whb, (long)GDIM * HDIM);
  cvt(out_W, outWb, (long)VOCAB * HDIM);

  // 2. metadata + biases + state init + embedding gather
  hipLaunchKernelGGL(meta_kernel, dim3(1), dim3(64), 0, stream, lengths, meta);
  hipLaunchKernelGGL(bias_sum_kernel, dim3((GDIM + 255) / 256), dim3(256), 0, stream,
                     b_ih, b_hh, bsum);
  hipLaunchKernelGGL(init_state_kernel, dim3(BATCH * HDIM / 256), dim3(256), 0, stream,
                     hcur, ccur, hb);
  hipLaunchKernelGGL(gather_embed_kernel,
                     dim3((BATCH * TCAP * EDIM) / 256), dim3(256), 0, stream,
                     caps, embed, Xb);

  // 3. visual projection -> Xb rows [0..B), bf16 output
  hipLaunchKernelGGL(gemm_wmma_kernel, dim3(BATCH / BM, EDIM / BN), dim3(256), 0, stream,
                     featb, (const __bf16*)nullptr, visWb, (const __bf16*)nullptr,
                     FDIM, visual_b, (float*)nullptr, Xb,
                     BATCH, EDIM, FDIM, (const int*)nullptr);

  // 4. recurrence: fused gates = x@W_ih^T + h@W_hh^T + bias, then cell
  for (int t = 0; t < TSEQ; ++t) {
    const __bf16* Xt = Xb + (size_t)t * BATCH * EDIM;
    hipLaunchKernelGGL(gemm_wmma_kernel, dim3(BATCH / BM, GDIM / BN), dim3(256), 0, stream,
                       Xt, hb, Wib, Whb, EDIM,
                       bsum, gates, (__bf16*)nullptr,
                       BATCH, GDIM, EDIM + HDIM, (const int*)nullptr);
    hipLaunchKernelGGL(lstm_cell_kernel, dim3(BATCH * HDIM / 256), dim3(256), 0, stream,
                       gates, hcur, ccur, hb, hsb, lengths, meta, dout, t);
  }

  // 5. logits over packed states: M bounded by meta[42] on device
  const int maxRows = TSEQ * BATCH;                        // 10496 upper bound
  hipLaunchKernelGGL(gemm_wmma_kernel,
                     dim3((maxRows + BM - 1) / BM, (VOCAB + BN - 1) / BN), dim3(256),
                     0, stream,
                     hsb, (const __bf16*)nullptr, outWb, (const __bf16*)nullptr,
                     HDIM, out_b, dout, (__bf16*)nullptr,
                     maxRows, VOCAB, HDIM, (const int*)&meta[42]);
}
